// LinearNO_Conv_2765958939225
// MI455X (gfx1250) — matrix operand order repaired and checked by reading, not compiled, not run on hardware
//
#include <hip/hip_runtime.h>
#include <hip/hip_bf16.h>
#include <cstdint>
#include <cstddef>

typedef __bf16 bf16;
typedef __attribute__((ext_vector_type(8)))  bf16  v8bf;
typedef __attribute__((ext_vector_type(16))) bf16  v16bf;
typedef __attribute__((ext_vector_type(8)))  float v8f;

#define B_     8
#define H_     128
#define W_     128
#define C_     256
#define N_     16384      /* H_*W_ */
#define HEADS_ 8
#define DH_    64
#define KR_    64
#define INNER_ 512

// ---------------------------------------------------------------------------
// WMMA bf16 16x16x32 fragment helpers (wave32, gfx1250).
// A (MxK=16x32): lane L holds row M=L&15; K chunks [hi*8,hi*8+8) and
//                [16+hi*8, +8), hi = L>>4.     (ISA 7.12.2, 16-bit A)
// B (KxN=32x16) from transposed storage Bt[n][k]: lane L holds col N=L&15;
//                K chunks [hi*8,hi*8+8) and [16+hi*8, +8) (same as A).
// C/D (16x16 f32): lane L -> col N=L&15; VGPR r -> row M = r + 8*hi.
// ---------------------------------------------------------------------------
__device__ __forceinline__ v16bf frag_a(const bf16* rowp, int hi) {
  v8bf a0 = *(const v8bf*)(rowp + hi * 8);
  v8bf a1 = *(const v8bf*)(rowp + 16 + hi * 8);
  v16bf r;
#pragma unroll
  for (int i = 0; i < 8; ++i) { r[i] = a0[i]; r[i + 8] = a1[i]; }
  return r;
}
__device__ __forceinline__ v16bf frag_b(const bf16* colp, int hi) {
  v8bf a0 = *(const v8bf*)(colp + hi * 8);
  v8bf a1 = *(const v8bf*)(colp + 16 + hi * 8);
  v16bf r;
#pragma unroll
  for (int i = 0; i < 8; ++i) { r[i] = a0[i]; r[i + 8] = a1[i]; }
  return r;
}
__device__ __forceinline__ v8f wmma_bf16(v16bf a, v16bf b, v8f c) {
  return __builtin_amdgcn_wmma_f32_16x16x32_bf16(false, a, false, b,
                                                 (short)0, c, false, false);
}

// ---------------------------------------------------------------------------
// gfx1250 async global->LDS copy (ISA 10.7 / 15.18.3), tracked by ASYNCcnt.
// ---------------------------------------------------------------------------
typedef __attribute__((address_space(3))) const void lds_cvoid;
__device__ __forceinline__ uint32_t lds_addr(const void* p) {
  return (uint32_t)(uintptr_t)(lds_cvoid*)p;
}
__device__ __forceinline__ void async_ld_b128(uint32_t lds, const void* gaddr) {
  asm volatile("global_load_async_to_lds_b128 %0, %1, off"
               :: "v"(lds), "v"(gaddr) : "memory");
}
__device__ __forceinline__ void wait_async0() {
  asm volatile("s_wait_asynccnt 0" ::: "memory");
}

// Cooperative copy of one contiguous 8 KB tile (128 rows x 32 bf16) into LDS.
// 512 x 16B chunks over 256 threads (2 async b128 per thread).
__device__ __forceinline__ void issue_tile8k(const bf16* __restrict__ g,
                                             bf16* lbuf, int tid) {
#pragma unroll
  for (int i = 0; i < 2; ++i) {
    const int chunk = tid + i * 256;
    async_ld_b128(lds_addr(lbuf + chunk * 8), g + (size_t)chunk * 8);
  }
}

// ---------------------------------------------------------------------------
// Pack kernels: fp32 -> bf16 in WMMA/async-tile friendly layouts.
// ---------------------------------------------------------------------------
__global__ void pack_bf16(const float* __restrict__ src, bf16* __restrict__ dst,
                          int n) {
  for (int i = blockIdx.x * blockDim.x + threadIdx.x; i < n;
       i += gridDim.x * blockDim.x)
    dst[i] = (bf16)src[i];
}

// conv_w [OC=512][IC=256][3][3] -> wp[tap*8+icc][oc][ici]  (8KB K-step tiles)
__global__ void pack_convw(const float* __restrict__ w, bf16* __restrict__ wp) {
  int g = blockIdx.x * blockDim.x + threadIdx.x;
  if (g >= 9 * 8 * INNER_ * 32) return;
  int ici = g & 31, rest = g >> 5;
  int oc = rest & 511, sic = rest >> 9;        // sic = tap*8 + icc
  int tap = sic >> 3, ic = (sic & 7) * 32 + ici;
  wp[g] = (bf16)w[(size_t)oc * (C_ * 9) + ic * 9 + tap];
}

// W [256][K] row-major -> Wp[kc][n][ki] (kc = K/32 chunks of 8KB per 128 cols)
__global__ void pack_bt32(const float* __restrict__ src, bf16* __restrict__ dst,
                          int K) {
  int g = blockIdx.x * blockDim.x + threadIdx.x;
  if (g >= 256 * K) return;
  int ki = g & 31, rest = g >> 5;
  int n = rest & 255, kc = rest >> 8;
  dst[g] = (bf16)src[(size_t)n * K + kc * 32 + ki];
}

// ---------------------------------------------------------------------------
// 3x3 conv as implicit GEMM.  Block = one image row (128 px) x 128 ocs,
// 8 waves, 72 K-steps (9 taps x 8 ic-chunks).  B tile double-buffered in LDS
// via async-to-LDS; A fragment register-prefetched one step ahead.
// ---------------------------------------------------------------------------
__global__ __launch_bounds__(256) void conv3x3_wmma(
    const bf16* __restrict__ xb, const bf16* __restrict__ wp,
    const float* __restrict__ cbias, bf16* __restrict__ xmid) {
  const int tid = threadIdx.x, wid = tid >> 5, lane = tid & 31;
  const int hi = lane >> 4, l15 = lane & 15;
  const int b = blockIdx.x >> 7, yrow = blockIdx.x & 127;
  const int oc0 = blockIdx.y * 128;
  const int mrow = wid * 16;

  __shared__ __align__(16) bf16 sB[2][128 * 32];

  auto issue = [&](int s) {
    issue_tile8k(wp + ((size_t)s * INNER_ + oc0) * 32, sB[s & 1], tid);
  };
  auto a_frag = [&](int s) -> v16bf {
    const int tap = s >> 3, icc = s & 7;
    const int dy = tap / 3 - 1, dx = tap % 3 - 1;
    const int sy = yrow + dy, px = mrow + l15 + dx;
    v16bf a = {};
    if (sy >= 0 && sy < H_ && px >= 0 && px < W_)
      a = frag_a(xb + (((size_t)b * H_ + sy) * W_ + px) * C_ + icc * 32, hi);
    return a;
  };

  v8f z = {};
  v8f acc[8] = {z, z, z, z, z, z, z, z};

  issue(0);
  v16bf a_cur = a_frag(0);
  for (int s = 0; s < 72; ++s) {
    wait_async0();             // our tile-s chunks have landed
    __syncthreads();           // everyone's landed; buf[(s+1)&1] free
    if (s + 1 < 72) issue(s + 1);
    v16bf a_nxt = (s + 1 < 72) ? a_frag(s + 1) : a_cur;
    const bf16* bbuf = sB[s & 1];
#pragma unroll
    for (int nt = 0; nt < 8; ++nt) {
      v16bf bf = frag_b(bbuf + (nt * 16 + l15) * 32, hi);
      acc[nt] = wmma_bf16(a_cur, bf, acc[nt]);
    }
    a_cur = a_nxt;
  }

#pragma unroll
  for (int nt = 0; nt < 8; ++nt) {
    const int oc = oc0 + nt * 16 + l15;
    const float bias = cbias[oc];
    const int head = oc >> 6, d = oc & 63;
    bf16* orow = xmid + (((size_t)b * HEADS_ + head) * N_) * DH_ + d;
#pragma unroll
    for (int r = 0; r < 8; ++r) {
      const int x = mrow + hi * 8 + r;
      orow[(size_t)(yrow * W_ + x) * DH_] = (bf16)(acc[nt][r] + bias);
    }
  }
}

// ---------------------------------------------------------------------------
// Attention pass 1 (per b,h and 128-row tile):
//   q = softmax_row(x @ Wq^T) -> qout;  e = exp(x @ Wk^T), v = x @ Wv^T (LDS,
//   transposed);  kvpart[tile] = e^T @ v;  cspart[tile] = colsums of e.
// A fragments are shared by the q/k/v GEMMs; B fragments preloaded in bulk.
// ---------------------------------------------------------------------------
__global__ __launch_bounds__(256) void attn_pass1(
    const bf16* __restrict__ xm, const bf16* __restrict__ wqb,
    const bf16* __restrict__ wkb, const bf16* __restrict__ wvb,
    bf16* __restrict__ qout, float* __restrict__ kvpart,
    float* __restrict__ cspart) {
  const int tid = threadIdx.x, wid = tid >> 5, lane = tid & 31;
  const int hi = lane >> 4, l15 = lane & 15;
  const int bh = blockIdx.y, tile = blockIdx.x;
  const int n0 = tile * 128;
  const bf16* xbase = xm + ((size_t)bh * N_ + n0) * DH_;

  __shared__ float s_f[8192];           // 32 KB, reused: qraw -> et|vt
  bf16* et = (bf16*)s_f;                // [KR_][128]
  bf16* vt = et + KR_ * 128;            // [DH_][128]

  v8f z = {};
  const bf16* arow = xbase + (size_t)(wid * 16 + l15) * DH_;
  const v16bf a0 = frag_a(arow, hi);
  const v16bf a1 = frag_a(arow + 32, hi);

  // ---- q raw GEMM (128x64, K=64): preload 8 B frags, then 8 WMMAs ----
  {
    v16bf bq[8];
#pragma unroll
    for (int kc = 0; kc < 2; ++kc)
#pragma unroll
      for (int nt = 0; nt < 4; ++nt)
        bq[kc * 4 + nt] =
            frag_b(wqb + (size_t)(nt * 16 + l15) * DH_ + kc * 32, hi);
    v8f cq[4] = {z, z, z, z};
#pragma unroll
    for (int nt = 0; nt < 4; ++nt) {
      cq[nt] = wmma_bf16(a0, bq[nt], cq[nt]);
      cq[nt] = wmma_bf16(a1, bq[4 + nt], cq[nt]);
    }
#pragma unroll
    for (int nt = 0; nt < 4; ++nt)
#pragma unroll
      for (int r = 0; r < 8; ++r)
        s_f[(wid * 16 + hi * 8 + r) * 64 + nt * 16 + l15] = cq[nt][r];
  }
  __syncthreads();
  // ---- row softmax over KR=64 ----
  if (tid < 128) {
    const float* row = s_f + tid * 64;
    float mx = -3.4e38f;
    for (int j = 0; j < 64; ++j) mx = fmaxf(mx, row[j]);
    float s = 0.f;
    for (int j = 0; j < 64; ++j) s += expf(row[j] - mx);
    const float inv = 1.0f / s;
    bf16* qrow = qout + ((size_t)bh * N_ + n0 + tid) * KR_;
    for (int j = 0; j < 64; ++j) qrow[j] = (bf16)(expf(row[j] - mx) * inv);
  }
  __syncthreads();

  // ---- k raw GEMM; e = exp(k) stored transposed et[kr][n_local] ----
  {
    v16bf bk[8];
#pragma unroll
    for (int kc = 0; kc < 2; ++kc)
#pragma unroll
      for (int nt = 0; nt < 4; ++nt)
        bk[kc * 4 + nt] =
            frag_b(wkb + (size_t)(nt * 16 + l15) * DH_ + kc * 32, hi);
    v8f ck[4] = {z, z, z, z};
#pragma unroll
    for (int nt = 0; nt < 4; ++nt) {
      ck[nt] = wmma_bf16(a0, bk[nt], ck[nt]);
      ck[nt] = wmma_bf16(a1, bk[4 + nt], ck[nt]);
    }
#pragma unroll
    for (int nt = 0; nt < 4; ++nt)
#pragma unroll
      for (int r = 0; r < 8; ++r)
        et[(nt * 16 + l15) * 128 + wid * 16 + hi * 8 + r] =
            (bf16)expf(ck[nt][r]);
  }
  // ---- v GEMM stored transposed vt[c][n_local] ----
  {
    v16bf bv[8];
#pragma unroll
    for (int kc = 0; kc < 2; ++kc)
#pragma unroll
      for (int nt = 0; nt < 4; ++nt)
        bv[kc * 4 + nt] =
            frag_b(wvb + (size_t)(nt * 16 + l15) * DH_ + kc * 32, hi);
    v8f cv[4] = {z, z, z, z};
#pragma unroll
    for (int nt = 0; nt < 4; ++nt) {
      cv[nt] = wmma_bf16(a0, bv[nt], cv[nt]);
      cv[nt] = wmma_bf16(a1, bv[4 + nt], cv[nt]);
    }
#pragma unroll
    for (int nt = 0; nt < 4; ++nt)
#pragma unroll
      for (int r = 0; r < 8; ++r)
        vt[(nt * 16 + l15) * 128 + wid * 16 + hi * 8 + r] = (bf16)cv[nt][r];
  }
  __syncthreads();

  // ---- per-tile column sums of e ----
  if (tid < 64) {
    float s = 0.f;
    for (int i = 0; i < 128; ++i) s += (float)et[tid * 128 + i];
    cspart[((size_t)bh * 128 + tile) * 64 + tid] = s;
  }
  // ---- kv partial = e^T @ v (64x64, K=128); 16 tiles over 8 waves ----
  {
    const int mt = wid >> 1, ntb = (wid & 1) * 2;
    v8f ckv[2] = {z, z};
    for (int kc = 0; kc < 4; ++kc) {
      v16bf a = frag_a(et + (mt * 16 + l15) * 128 + kc * 32, hi);
#pragma unroll
      for (int j = 0; j < 2; ++j) {
        v16bf bf = frag_b(vt + ((ntb + j) * 16 + l15) * 128 + kc * 32, hi);
        ckv[j] = wmma_bf16(a, bf, ckv[j]);
      }
    }
    float* kvp = kvpart + ((size_t)bh * 128 + tile) * 4096;
#pragma unroll
    for (int j = 0; j < 2; ++j)
#pragma unroll
      for (int r = 0; r < 8; ++r)
        kvp[(mt * 16 + hi * 8 + r) * 64 + (ntb + j) * 16 + l15] = ckv[j][r];
  }
}

// colsum[bh][kr] = sum over 128 tiles of cspart (deterministic)
__global__ void reduce_colsum(const float* __restrict__ cspart,
                              float* __restrict__ colsum) {
  int g = blockIdx.x * blockDim.x + threadIdx.x;
  if (g >= 64 * 64) return;
  int bh = g >> 6, kr = g & 63;
  const float* p = cspart + ((size_t)bh * 128) * 64 + kr;
  float s = 0.f;
  for (int t = 0; t < 128; ++t) s += p[(size_t)t * 64];
  colsum[g] = s;
}

// kvT[bh][c][kr] = (sum_tiles kvpart[bh][tile][kr][c]) / colsum[bh][kr]
__global__ void kv_finalize(const float* __restrict__ kvpart,
                            const float* __restrict__ colsum,
                            bf16* __restrict__ kvt) {
  int g = blockIdx.x * blockDim.x + threadIdx.x;
  if (g >= 64 * 4096) return;
  int bh = g >> 12, r = g & 4095, kr = r >> 6, c = r & 63;
  const float* p = kvpart + ((size_t)bh * 128) * 4096 + r;
  float acc = 0.f;
  for (int t = 0; t < 128; ++t) acc += p[(size_t)t * 4096];
  kvt[((size_t)bh << 12) + c * 64 + kr] = (bf16)(acc / colsum[(bh << 6) + kr]);
}

// ---------------------------------------------------------------------------
// Attention pass 3: qkv = q @ kv (128x64 per block, K=64) -> ao[b][n][h*64+c]
// ---------------------------------------------------------------------------
__global__ __launch_bounds__(256) void attn_pass3(
    const bf16* __restrict__ q, const bf16* __restrict__ kvt,
    bf16* __restrict__ ao) {
  const int tid = threadIdx.x, wid = tid >> 5, lane = tid & 31;
  const int hi = lane >> 4, l15 = lane & 15;
  const int bh = blockIdx.y, n0 = blockIdx.x * 128;
  const int b = bh >> 3, hh = bh & 7;
  const bf16* kvb = kvt + ((size_t)bh << 12);

  const bf16* arow = q + ((size_t)bh * N_ + n0 + wid * 16 + l15) * KR_;
  const v16bf a0 = frag_a(arow, hi);
  const v16bf a1 = frag_a(arow + 32, hi);
  v16bf bb[8];
#pragma unroll
  for (int kc = 0; kc < 2; ++kc)
#pragma unroll
    for (int nt = 0; nt < 4; ++nt)
      bb[kc * 4 + nt] = frag_b(kvb + (size_t)(nt * 16 + l15) * KR_ + kc * 32, hi);

  v8f z = {};
  v8f cc[4] = {z, z, z, z};
#pragma unroll
  for (int nt = 0; nt < 4; ++nt) {
    cc[nt] = wmma_bf16(a0, bb[nt], cc[nt]);
    cc[nt] = wmma_bf16(a1, bb[4 + nt], cc[nt]);
  }
#pragma unroll
  for (int nt = 0; nt < 4; ++nt)
#pragma unroll
    for (int r = 0; r < 8; ++r) {
      const int m = wid * 16 + hi * 8 + r;
      ao[((size_t)b * N_ + n0 + m) * INNER_ + hh * 64 + nt * 16 + l15] =
          (bf16)cc[nt][r];
    }
}

// ---------------------------------------------------------------------------
// 128x128-tile GEMM, B in [kc][n][32] layout, async double-buffered via LDS.
// mode 0: GELU(exact) epilogue -> bf16;  mode 1: +bias -> f32.
// ---------------------------------------------------------------------------
__global__ __launch_bounds__(256) void gemm128(
    const bf16* __restrict__ A, int lda, const bf16* __restrict__ Bp, int K,
    const float* __restrict__ bias, void* __restrict__ out, int mode) {
  const int tid = threadIdx.x, wid = tid >> 5, lane = tid & 31;
  const int hi = lane >> 4, l15 = lane & 15;
  const int mb = blockIdx.x * 128, cb = blockIdx.y * 128;
  const int steps = K >> 5;

  __shared__ __align__(16) bf16 sB[2][128 * 32];
  auto issue = [&](int s) {
    issue_tile8k(Bp + ((size_t)s * 256 + cb) * 32, sB[s & 1], tid);
  };

  const bf16* arow = A + (size_t)(mb + wid * 16 + l15) * lda;
  v8f z = {};
  v8f cc[8] = {z, z, z, z, z, z, z, z};

  issue(0);
  v16bf a_cur = frag_a(arow, hi);
  for (int s = 0; s < steps; ++s) {
    wait_async0();
    __syncthreads();
    if (s + 1 < steps) issue(s + 1);
    v16bf a_nxt = (s + 1 < steps) ? frag_a(arow + (s + 1) * 32, hi) : a_cur;
    const bf16* bbuf = sB[s & 1];
#pragma unroll
    for (int nt = 0; nt < 8; ++nt) {
      v16bf bf = frag_b(bbuf + (nt * 16 + l15) * 32, hi);
      cc[nt] = wmma_bf16(a_cur, bf, cc[nt]);
    }
    a_cur = a_nxt;
  }

  if (mode == 0) {
#pragma unroll
    for (int nt = 0; nt < 8; ++nt) {
      const int col = cb + nt * 16 + l15;
      const float bv = bias[col];
#pragma unroll
      for (int r = 0; r < 8; ++r) {
        const size_t row = (size_t)mb + wid * 16 + hi * 8 + r;
        float v = cc[nt][r] + bv;
        v = 0.5f * v * (1.0f + erff(v * 0.70710678118654752f));
        ((bf16*)out)[row * 256 + col] = (bf16)v;
      }
    }
  } else {
#pragma unroll
    for (int nt = 0; nt < 8; ++nt) {
      const int col = cb + nt * 16 + l15;
      const float bv = bias[col];
#pragma unroll
      for (int r = 0; r < 8; ++r) {
        const size_t row = (size_t)mb + wid * 16 + hi * 8 + r;
        ((float*)out)[row * 256 + col] = cc[nt][r] + bv;
      }
    }
  }
}

// ---------------------------------------------------------------------------
extern "C" void kernel_launch(void* const* d_in, const int* in_sizes, int n_in,
                              void* d_out, int out_size, void* d_ws,
                              size_t ws_size, hipStream_t stream) {
  (void)in_sizes; (void)n_in; (void)out_size; (void)ws_size;

  const float* x      = (const float*)d_in[0];
  const float* conv_w = (const float*)d_in[1];
  const float* conv_b = (const float*)d_in[2];
  const float* Wq     = (const float*)d_in[3];
  const float* Wk     = (const float*)d_in[4];
  const float* Wv     = (const float*)d_in[5];
  const float* W1     = (const float*)d_in[6];
  const float* b1     = (const float*)d_in[7];
  const float* W2     = (const float*)d_in[8];
  const float* b2     = (const float*)d_in[9];

  uint8_t* ws = (uint8_t*)d_ws;
  size_t off = 0;
  auto alloc = [&](size_t bytes) -> uint8_t* {
    uint8_t* p = ws + off;
    off += (bytes + 255) & ~(size_t)255;
    return p;
  };

  const size_t nx   = (size_t)B_ * N_ * C_;             // 33.55 M
  const size_t nmid = (size_t)B_ * HEADS_ * N_ * DH_;   // 67.1 M

  bf16*  xb     = (bf16*)alloc(nx * 2);
  bf16*  wp     = (bf16*)alloc((size_t)9 * 8 * INNER_ * 32 * 2);
  bf16*  wqb    = (bf16*)alloc(KR_ * DH_ * 2);
  bf16*  wkb    = (bf16*)alloc(KR_ * DH_ * 2);
  bf16*  wvb    = (bf16*)alloc(DH_ * DH_ * 2);
  bf16*  w1p    = (bf16*)alloc((size_t)16 * 256 * 32 * 2);
  bf16*  w2p    = (bf16*)alloc((size_t)8 * 256 * 32 * 2);
  bf16*  xmid   = (bf16*)alloc(nmid * 2);               // reused as ao
  bf16*  q      = (bf16*)alloc(nmid * 2);
  float* kvpart = (float*)alloc((size_t)64 * 128 * 4096 * 4);
  float* cspart = (float*)alloc((size_t)64 * 128 * 64 * 4);
  float* colsum = (float*)alloc((size_t)64 * 64 * 4);
  bf16*  kvt    = (bf16*)alloc((size_t)64 * 4096 * 2);
  bf16*  hbuf   = (bf16*)alloc(nx * 2);
  bf16*  ao     = xmid;   // x_mid dead after pass1; alias for qkv output

  // 1) downcast packs
  pack_bf16<<<16384, 256, 0, stream>>>(x, xb, (int)nx);
  pack_bf16<<<16, 256, 0, stream>>>(Wq, wqb, KR_ * DH_);
  pack_bf16<<<16, 256, 0, stream>>>(Wk, wkb, KR_ * DH_);
  pack_bf16<<<16, 256, 0, stream>>>(Wv, wvb, DH_ * DH_);
  pack_bt32<<<(256 * 512 + 255) / 256, 256, 0, stream>>>(W1, w1p, INNER_);
  pack_bt32<<<(256 * 256 + 255) / 256, 256, 0, stream>>>(W2, w2p, C_);
  pack_convw<<<(9 * 8 * INNER_ * 32 + 255) / 256, 256, 0, stream>>>(conv_w, wp);

  // 2) 3x3 conv as implicit WMMA GEMM (async B tiles)
  conv3x3_wmma<<<dim3(B_ * H_, INNER_ / 128), 256, 0, stream>>>(
      xb, wp, conv_b, xmid);

  // 3) attention
  attn_pass1<<<dim3(N_ / 128, B_ * HEADS_), 256, 0, stream>>>(
      xmid, wqb, wkb, wvb, q, kvpart, cspart);
  reduce_colsum<<<(64 * 64 + 255) / 256, 256, 0, stream>>>(cspart, colsum);
  kv_finalize<<<(64 * 4096 + 255) / 256, 256, 0, stream>>>(kvpart, colsum,
                                                           kvt);
  attn_pass3<<<dim3(N_ / 128, B_ * HEADS_), 256, 0, stream>>>(q, kvt, ao);

  // 4) MLP: GELU(ao @ W1^T + b1) @ W2^T + b2
  gemm128<<<dim3((B_ * N_) / 128, C_ / 128), 256, 0, stream>>>(
      ao, INNER_, w1p, INNER_, b1, hbuf, 0);
  gemm128<<<dim3((B_ * N_) / 128, C_ / 128), 256, 0, stream>>>(
      hbuf, C_, w2p, C_, b2, d_out, 1);
}
